// Encoder_38637525795020
// MI455X (gfx1250) — compile-verified, matrix operand order checked by reading, not run on hardware
//
#include <hip/hip_runtime.h>

#define HH   128
#define WW   128
#define HWP  16384          // 128*128
#define NBATCH 4
#define NPIX (NBATCH*HWP)   // 65536 output pixels
#define TN   128            // pixel tile per workgroup

typedef _Float16 v16h __attribute__((ext_vector_type(16)));
typedef _Float16 v8h  __attribute__((ext_vector_type(8)));
typedef float    v8f  __attribute__((ext_vector_type(8)));

__device__ __forceinline__ v16h cat16(v8h lo, v8h hi) {
  return __builtin_shufflevector(lo, hi, 0,1,2,3,4,5,6,7,8,9,10,11,12,13,14,15);
}

// ---------------------------------------------------------------- maxpool 2x2
__global__ void maxpool2_k(const float* __restrict__ in, float* __restrict__ out, int total) {
  int i = blockIdx.x * 256 + threadIdx.x;
  if (i >= total) return;
  int x  = i & (WW - 1);
  int y  = (i >> 7) & (HH - 1);
  int bc = i >> 14;
  const float* p = in + ((size_t)bc * 256 + 2 * y) * 256 + 2 * x;
  out[i] = fmaxf(fmaxf(p[0], p[1]), fmaxf(p[256], p[257]));
}

// ------------------------------------------------- pack f32 OIHW -> f16 O x Kpad
__global__ void pack_w_k(const float* __restrict__ w, const float* __restrict__ b,
                         _Float16* __restrict__ wout, float* __restrict__ bout,
                         int O, int Opad, int K, int Kpad) {
  int i = blockIdx.x * 256 + threadIdx.x;
  if (i < Opad * Kpad) {
    int o = i / Kpad, k = i - o * Kpad;
    wout[i] = (_Float16)((o < O && k < K) ? w[o * K + k] : 0.f);
  }
  if (i < Opad) bout[i] = (i < O) ? b[i] : 0.f;
}

// ---------------------------------------------------------------- channel concat
__global__ void cat_k(const float* __restrict__ a, const float* __restrict__ b,
                      const float* __restrict__ c, float* __restrict__ out, int total) {
  int i = blockIdx.x * 256 + threadIdx.x;
  if (i >= total) return;
  int pix = i & (HWP - 1);
  int t   = i >> 14;
  int ch  = t % 192;
  int bb  = t / 192;
  float v;
  if (ch < 64)       v = a[((size_t)(bb * 64 + ch))       * HWP + pix];
  else if (ch < 128) v = b[((size_t)(bb * 64 + ch - 64))  * HWP + pix];
  else               v = c[((size_t)(bb * 64 + ch - 128)) * HWP + pix];
  out[i] = v;
}

// --------------------------------------------- implicit-GEMM conv via WMMA f16
// out tile: 64 channels x 128 pixels per workgroup; 8 wave32, each wave owns a
// 16(M) x 64(N) sub-tile = 4 f32 accumulators, 4 WMMAs per K-step per wave.
__global__ __launch_bounds__(256)
void conv_wmma_k(const float* __restrict__ x, const _Float16* __restrict__ wpk,
                 const float* __restrict__ bias, const float* __restrict__ res,
                 float* __restrict__ out,
                 int Cin, int K, int Kpad, int ksize, int relu)
{
  __shared__ _Float16 lsA[64][48] __attribute__((aligned(16)));
  __shared__ _Float16 lsB[TN][48] __attribute__((aligned(16)));

  const int tid   = threadIdx.x;
  const int lane  = tid & 31;
  const int wid   = tid >> 5;
  const int waveM = wid & 3;     // 4 M sub-tiles of 16
  const int waveN = wid >> 2;    // 2 N sub-tiles of 64
  const int n0    = blockIdx.x * TN;
  const int moff  = blockIdx.y * 64;

  const int kper = ksize * ksize;
  const int pd   = (ksize == 3) ? 1 : 0;

  // A fill: 64x32 halves, 8 per thread
  const int arow = tid >> 2;
  const int acol = (tid & 3) * 8;
  // B fill: TN x 32 halves; thread -> pixel (coalesced loads), 2 chunks of 8 K
  const int bn     = tid & (TN - 1);
  const int bkbase = (tid >> 7) * 16;
  const int bng    = n0 + bn;
  const int bb     = bng >> 14;
  const int brem   = bng & (HWP - 1);
  const int by     = brem >> 7;
  const int bx     = brem & (WW - 1);

  v8f acc[4];
#pragma unroll
  for (int q = 0; q < 4; ++q) acc[q] = (v8f){};

  for (int k0 = 0; k0 < Kpad; k0 += 32) {
    // stage A (weights, already f16) + prefetch next K slice
    *(v8h*)&lsA[arow][acol] =
        *(const v8h*)(wpk + (size_t)(moff + arow) * Kpad + k0 + acol);
    if (k0 + 32 < Kpad)
      __builtin_prefetch(wpk + (size_t)(moff + arow) * Kpad + k0 + 32 + acol, 0, 0);

    // stage B (im2col patches, f32 -> f16), one b128 LDS store per 8-K chunk
#pragma unroll
    for (int g = 0; g < 2; ++g) {
      const int kb = bkbase + g * 8;
      int k   = k0 + kb;
      int c   = k / kper;
      int kid = k - c * kper;
      v8h vb;
#pragma unroll
      for (int i = 0; i < 8; ++i) {
        float v = 0.f;
        if (k + i < K) {
          int ky = kid / ksize - pd;
          int kx = kid - (kid / ksize) * ksize - pd;
          int yy = by + ky, xs = bx + kx;
          if (yy >= 0 && yy < HH && xs >= 0 && xs < WW)
            v = x[(size_t)(bb * Cin + c) * HWP + yy * WW + xs];
        }
        vb[i] = (_Float16)v;
        if (++kid == kper) { kid = 0; ++c; }
      }
      *(v8h*)&lsB[bn][kb] = vb;
    }
    __syncthreads();

    const int r  = lane & 15;
    const int b8 = (lane >> 4) * 8;   // lane<16: K 0..7/16..23; lane>=16: +8
    v16h af = cat16(*(const v8h*)&lsA[waveM * 16 + r][b8],
                    *(const v8h*)&lsA[waveM * 16 + r][b8 + 16]);
#pragma unroll
    for (int q = 0; q < 4; ++q) {
      const int col = waveN * 64 + q * 16 + r;
      v16h bf = cat16(*(const v8h*)&lsB[col][b8],
                      *(const v8h*)&lsB[col][b8 + 16]);
      acc[q] = __builtin_amdgcn_wmma_f32_16x16x32_f16(false, af, false, bf,
                                                      (short)0, acc[q], false, false);
    }
    __syncthreads();
  }

  // epilogue: C/D layout -> M = e + 8*(lane>=16), N = lane&15
  const int ncol = lane & 15;
  const int mhi  = (lane >> 4) * 8;
#pragma unroll
  for (int q = 0; q < 4; ++q) {
    int ng  = n0 + waveN * 64 + q * 16 + ncol;
    int b   = ng >> 14;
    int rem = ng & (HWP - 1);
#pragma unroll
    for (int e = 0; e < 8; ++e) {
      int ch = moff + waveM * 16 + mhi + e;
      size_t idx = (size_t)(b * 64 + ch) * HWP + rem;
      float v = acc[q][e] + bias[ch];
      if (res)  v += res[idx];
      if (relu) v = fmaxf(v, 0.f);
      out[idx] = v;
    }
  }
}

// ------------------------------- deformable conv: bilinear gather + WMMA GEMM
__global__ __launch_bounds__(256)
void dcn_wmma_k(const float* __restrict__ x, const float* __restrict__ ob,
                const _Float16* __restrict__ wpk, const float* __restrict__ bias,
                float* __restrict__ out)
{
  __shared__ _Float16 lsA[64][48] __attribute__((aligned(16)));
  __shared__ _Float16 lsB[TN][48] __attribute__((aligned(16)));
  __shared__ int   cIx[9][TN][4];
  __shared__ float cWt[9][TN][4];

  const int tid   = threadIdx.x;
  const int lane  = tid & 31;
  const int wid   = tid >> 5;
  const int waveM = wid & 3;
  const int waveN = wid >> 2;
  const int n0    = blockIdx.x * TN;

  // phase 0: per-pixel/per-tap bilinear corners, weight folded with sigmoid(mask)
  for (int e = tid; e < 9 * TN; e += 256) {
    int kidx = e >> 7;
    int n    = e & (TN - 1);
    int ng   = n0 + n;
    int b    = ng >> 14;
    int rem  = ng & (HWP - 1);
    int y = rem >> 7, xx = rem & (WW - 1);
    size_t base = (size_t)(b * 64 + kidx) * HWP + rem;
    float dy = ob[base];                 // ch kidx
    float dx = ob[base + 9  * HWP];      // ch 9+kidx
    float mm = ob[base + 18 * HWP];      // ch 18+kidx
    mm = 1.f / (1.f + __expf(-mm));
    float py = (float)(y  + kidx / 3 - 1) + dy;
    float px = (float)(xx + kidx % 3 - 1) + dx;
    float y0f = floorf(py), x0f = floorf(px);
    float wy1 = py - y0f,  wx1 = px - x0f;
#pragma unroll
    for (int j = 0; j < 4; ++j) {
      float yc = y0f + (float)(j >> 1);
      float xc = x0f + (float)(j & 1);
      float wgt = ((j >> 1) ? wy1 : 1.f - wy1) * ((j & 1) ? wx1 : 1.f - wx1);
      bool valid = (yc >= 0.f) && (yc <= 127.f) && (xc >= 0.f) && (xc <= 127.f);
      int yi = (int)fminf(fmaxf(yc, 0.f), 127.f);
      int xi = (int)fminf(fmaxf(xc, 0.f), 127.f);
      cIx[kidx][n][j] = b * 64 * HWP + yi * WW + xi;   // add c*HWP later
      cWt[kidx][n][j] = valid ? wgt * mm : 0.f;
    }
  }
  __syncthreads();

  const int arow = tid >> 2;
  const int acol = (tid & 3) * 8;
  const int bn     = tid & (TN - 1);
  const int bkbase = (tid >> 7) * 16;

  v8f acc[4];
#pragma unroll
  for (int q = 0; q < 4; ++q) acc[q] = (v8f){};

  for (int k0 = 0; k0 < 576; k0 += 32) {
    *(v8h*)&lsA[arow][acol] = *(const v8h*)(wpk + (size_t)arow * 576 + k0 + acol);
    if (k0 + 32 < 576)
      __builtin_prefetch(wpk + (size_t)arow * 576 + k0 + 32 + acol, 0, 0);

#pragma unroll
    for (int g = 0; g < 2; ++g) {
      const int kb = bkbase + g * 8;
      int k   = k0 + kb;
      int c   = k / 9;
      int kid = k - c * 9;
      v8h vb;
#pragma unroll
      for (int i = 0; i < 8; ++i) {
        int coff = c * HWP;
        float v = cWt[kid][bn][0] * x[cIx[kid][bn][0] + coff]
                + cWt[kid][bn][1] * x[cIx[kid][bn][1] + coff]
                + cWt[kid][bn][2] * x[cIx[kid][bn][2] + coff]
                + cWt[kid][bn][3] * x[cIx[kid][bn][3] + coff];
        vb[i] = (_Float16)v;
        if (++kid == 9) { kid = 0; ++c; }
      }
      *(v8h*)&lsB[bn][kb] = vb;
    }
    __syncthreads();

    const int r  = lane & 15;
    const int b8 = (lane >> 4) * 8;
    v16h af = cat16(*(const v8h*)&lsA[waveM * 16 + r][b8],
                    *(const v8h*)&lsA[waveM * 16 + r][b8 + 16]);
#pragma unroll
    for (int q = 0; q < 4; ++q) {
      const int col = waveN * 64 + q * 16 + r;
      v16h bf = cat16(*(const v8h*)&lsB[col][b8],
                      *(const v8h*)&lsB[col][b8 + 16]);
      acc[q] = __builtin_amdgcn_wmma_f32_16x16x32_f16(false, af, false, bf,
                                                      (short)0, acc[q], false, false);
    }
    __syncthreads();
  }

  const int ncol = lane & 15;
  const int mhi  = (lane >> 4) * 8;
#pragma unroll
  for (int q = 0; q < 4; ++q) {
    int ng  = n0 + waveN * 64 + q * 16 + ncol;
    int b   = ng >> 14;
    int rem = ng & (HWP - 1);
#pragma unroll
    for (int e = 0; e < 8; ++e) {
      int ch = waveM * 16 + mhi + e;
      out[(size_t)(b * 64 + ch) * HWP + rem] = acc[q][e] + bias[ch];
    }
  }
}

// ---------------------------------------------------------------- host driver
extern "C" void kernel_launch(void* const* d_in, const int* in_sizes, int n_in,
                              void* d_out, int out_size, void* d_ws, size_t ws_size,
                              hipStream_t stream)
{
  (void)in_sizes; (void)n_in; (void)out_size; (void)ws_size;
  const float* xl   = (const float*)d_in[0];
  const float* xr   = (const float*)d_in[1];
  const float* cv   = (const float*)d_in[2];
  const float* cbw  = (const float*)d_in[3];  const float* cbb  = (const float*)d_in[4];
  const float* r11w = (const float*)d_in[5];  const float* r11b = (const float*)d_in[6];
  const float* r12w = (const float*)d_in[7];  const float* r12b = (const float*)d_in[8];
  const float* r21w = (const float*)d_in[9];  const float* r21b = (const float*)d_in[10];
  const float* r22w = (const float*)d_in[11]; const float* r22b = (const float*)d_in[12];
  const float* f1w  = (const float*)d_in[13]; const float* f1b  = (const float*)d_in[14];
  const float* f2w  = (const float*)d_in[15]; const float* f2b  = (const float*)d_in[16];
  const float* olw  = (const float*)d_in[17]; const float* olbv = (const float*)d_in[18];
  const float* orw  = (const float*)d_in[19]; const float* orbv = (const float*)d_in[20];
  const float* dwv  = (const float*)d_in[21]; const float* dbv  = (const float*)d_in[22];

  const size_t A64  = (size_t)NBATCH * 64  * HWP;
  const size_t A81  = (size_t)NBATCH * 81  * HWP;
  const size_t A192 = (size_t)NBATCH * 192 * HWP;

  char* p = (char*)d_ws;
  auto alloc_f = [&](size_t n) { float* r = (float*)p; p += n * sizeof(float); return r; };
  auto alloc_h = [&](size_t n) { _Float16* r = (_Float16*)p; p += n * sizeof(_Float16); return r; };

  float* poolb = alloc_f(A64);   // pooled input (reused xl then xr)
  float* cvp   = alloc_f(A81);
  float* c1    = alloc_f(A64);
  float* c2    = alloc_f(A64);
  float* c3    = alloc_f(A64);
  float* xlb   = alloc_f(A64);
  float* xrb   = alloc_f(A64);
  float* catb  = alloc_f(A192);

  _Float16* Wcb = alloc_h(64 * 576);
  _Float16* W11 = alloc_h(64 * 576);
  _Float16* W12 = alloc_h(64 * 576);
  _Float16* W21 = alloc_h(64 * 576);
  _Float16* W22 = alloc_h(64 * 576);
  _Float16* Wf1 = alloc_h(64 * 96);
  _Float16* Wf2 = alloc_h(64 * 64);
  _Float16* Wol = alloc_h(64 * 1728);
  _Float16* Wor = alloc_h(64 * 1728);
  _Float16* Wd  = alloc_h(64 * 576);
  float* Bcb = alloc_f(64); float* B11 = alloc_f(64); float* B12 = alloc_f(64);
  float* B21 = alloc_f(64); float* B22 = alloc_f(64); float* Bf1 = alloc_f(64);
  float* Bf2 = alloc_f(64); float* Bol = alloc_f(64); float* Bor = alloc_f(64);
  float* Bd  = alloc_f(64);

  dim3 blk(256);
  auto pack = [&](const float* w, const float* b, _Float16* wo, float* bo,
                  int O, int Opad, int K, int Kpad) {
    int tot = Opad * Kpad;
    pack_w_k<<<dim3((tot + 255) / 256), blk, 0, stream>>>(w, b, wo, bo, O, Opad, K, Kpad);
  };
  pack(cbw,  cbb,  Wcb, Bcb, 64, 64, 576,  576);
  pack(r11w, r11b, W11, B11, 64, 64, 576,  576);
  pack(r12w, r12b, W12, B12, 64, 64, 576,  576);
  pack(r21w, r21b, W21, B21, 64, 64, 576,  576);
  pack(r22w, r22b, W22, B22, 64, 64, 576,  576);
  pack(f1w,  f1b,  Wf1, Bf1, 64, 64, 81,   96);
  pack(f2w,  f2b,  Wf2, Bf2, 64, 64, 64,   64);
  pack(olw,  olbv, Wol, Bol, 27, 64, 1728, 1728);
  pack(orw,  orbv, Wor, Bor, 27, 64, 1728, 1728);
  pack(dwv,  dbv,  Wd,  Bd,  64, 64, 576,  576);

  dim3 cgrid(NPIX / TN, 1);
  auto conv = [&](const float* in, const _Float16* w, const float* b, const float* res,
                  float* o, int Cin, int K, int Kpad, int ks, int relu) {
    conv_wmma_k<<<cgrid, blk, 0, stream>>>(in, w, b, res, o, Cin, K, Kpad, ks, relu);
  };

  // ---- left tower
  maxpool2_k<<<dim3(((int)A64 + 255) / 256), blk, 0, stream>>>(xl, poolb, (int)A64);
  conv(poolb, Wcb, Bcb, nullptr, c1, 64, 576, 576, 3, 1);   // cb + relu
  conv(c1,    W11, B11, nullptr, c2, 64, 576, 576, 3, 1);   // r11 + relu
  conv(c2,    W12, B12, c1,      c3, 64, 576, 576, 3, 0);   // r12 + residual
  conv(c3,    W21, B21, nullptr, c1, 64, 576, 576, 3, 1);   // r21 + relu
  conv(c1,    W22, B22, c3,      xlb, 64, 576, 576, 3, 0);  // r22 + residual

  // ---- right tower
  maxpool2_k<<<dim3(((int)A64 + 255) / 256), blk, 0, stream>>>(xr, poolb, (int)A64);
  conv(poolb, Wcb, Bcb, nullptr, c1, 64, 576, 576, 3, 1);
  conv(c1,    W11, B11, nullptr, c2, 64, 576, 576, 3, 1);
  conv(c2,    W12, B12, c1,      c3, 64, 576, 576, 3, 0);
  conv(c3,    W21, B21, nullptr, c1, 64, 576, 576, 3, 1);
  conv(c1,    W22, B22, c3,      xrb, 64, 576, 576, 3, 0);

  // ---- cost-volume tower (1x1 convs)
  maxpool2_k<<<dim3(((int)A81 + 255) / 256), blk, 0, stream>>>(cv, cvp, (int)A81);
  conv(cvp, Wf1, Bf1, nullptr, c1, 81, 81, 96, 1, 1);
  conv(c1,  Wf2, Bf2, nullptr, c2, 64, 64, 64, 1, 1);

  // ---- concat + offset convs
  cat_k<<<dim3(((int)A192 + 255) / 256), blk, 0, stream>>>(xlb, xrb, c2, catb, (int)A192);
  conv(catb, Wol, Bol, nullptr, c3, 192, 1728, 1728, 3, 0); // ol (27ch in 64-pad)
  conv(catb, Wor, Bor, nullptr, c1, 192, 1728, 1728, 3, 0); // or

  // ---- deformable convs -> outputs
  float* yl = (float*)d_out;
  float* yr = yl + A64;
  dcn_wmma_k<<<cgrid, blk, 0, stream>>>(xlb, c3, Wd, Bd, yl);
  dcn_wmma_k<<<cgrid, blk, 0, stream>>>(xrb, c1, Wd, Bd, yr);
}
